// ONOBlock_63333587747478
// MI455X (gfx1250) — matrix operand order repaired and checked by reading, not compiled, not run on hardware
//
#include <hip/hip_runtime.h>
#include <math.h>

// ---------------------------------------------------------------------------
// Types for CDNA5 WMMA (gfx1250, wave32)
// ---------------------------------------------------------------------------
typedef __bf16 bf16_t;
typedef __attribute__((ext_vector_type(8)))  __bf16 v8bf;
typedef __attribute__((ext_vector_type(16))) __bf16 v16bf;
typedef __attribute__((ext_vector_type(8)))  float  v8f;
typedef __attribute__((ext_vector_type(4)))  int    v4i;
typedef __attribute__((ext_vector_type(8)))  int    v8i;

#define TILE  128
#define KSTEP 32
#define LDP   40   // LDS row pitch (elements); 80B keeps 16B alignment, kills phase conflicts

__device__ inline void wait_tensorcnt0() {
#if __has_builtin(__builtin_amdgcn_s_wait_tensorcnt)
    __builtin_amdgcn_s_wait_tensorcnt(0);
#else
    asm volatile("s_wait_tensorcnt 0x0" ::: "memory");
#endif
}

// ---------------------------------------------------------------------------
// Tensor Data Mover: async 2D tile load global->LDS (CDNA5 ISA)
// D# group0/group1 built in SGPRs; inline asm is portable across both
// toolchains (bypasses the 5-arg vs 6-arg builtin arity split).
// Loads tile_rows x tile_k bf16 elements (row stride row_stride_elems) into
// LDS at lds_addr, padding each 64B row by 16B -> LDS pitch = 80B = LDP*2.
// ---------------------------------------------------------------------------
__device__ inline void tdm_load_tile_bf16(unsigned lds_addr, const void* gaddr,
                                          unsigned tile_k, unsigned tile_rows,
                                          long row_stride_elems)
{
    const unsigned long long ga = (unsigned long long)(uintptr_t)gaddr;
    const unsigned long long st = (unsigned long long)row_stride_elems;
    v4i g0; v8i g1;
    g0[0] = 1;                                              // count=1, user D#
    g0[1] = (int)lds_addr;                                  // lds_addr
    g0[2] = (int)(ga & 0xffffffffu);                        // global_addr[31:0]
    g0[3] = (int)(((ga >> 32) & 0x1ffffffu) | (2u << 30));  // addr[56:32] | type=2
    // data_size=2B (code 1), pad_enable, pad_interval=16dw (code 3),
    // pad_amount=4dw (code 3), workgroup_mask=0 (not in cluster)
    g1[0] = (1 << 16) | (1 << 20) | (3 << 22) | (3 << 25);
    g1[1] = (int)((tile_k & 0xffffu) << 16);                // tensor_dim0[15:0]
    g1[2] = (int)(((tile_k >> 16) & 0xffffu) |              // tensor_dim0[31:16]
                  ((tile_rows & 0xffffu) << 16));           // tensor_dim1[15:0]
    g1[3] = (int)(((tile_rows >> 16) & 0xffffu) |           // tensor_dim1[31:16]
                  ((tile_k & 0xffffu) << 16));              // tile_dim0
    g1[4] = (int)(tile_rows & 0xffffu);                     // tile_dim1 (tile_dim2=0)
    g1[5] = (int)(st & 0xffffffffu);                        // tensor_dim0_stride[31:0]
    g1[6] = (int)((st >> 32) & 0xffffu);                    // stride[47:32], dim1_stride=0
    g1[7] = 0;
    asm volatile("tensor_load_to_lds %0, %1" :: "s"(g0), "s"(g1) : "memory");
}

// ---------------------------------------------------------------------------
// Fragment loaders per CDNA5 ISA 7.12.2 (wave32)
// A 16x32 bf16: lanes 0-15 hold row M=lane, K 0..7 (e0-7) and 16..23 (e8-15);
//               lanes 16-31 hold K 8..15 and 24..31.
// B 32x16 bf16 (tile stored N-major in LDS): lane<16 col=lane K 0..7 and 16..23,
//               lane>=16 col=lane-16 K 8..15 and 24..31 (same K order as A).
// ---------------------------------------------------------------------------
__device__ inline v16bf load_frag_a(const bf16_t* base, int lane) {
    const bf16_t* p = base + (lane & 15) * LDP + ((lane >> 4) * 8);
    v8bf lo = *(const v8bf*)p;
    v8bf hi = *(const v8bf*)(p + 16);
    return __builtin_shufflevector(lo, hi, 0,1,2,3,4,5,6,7,8,9,10,11,12,13,14,15);
}
__device__ inline v16bf load_frag_b(const bf16_t* base, int lane) {
    const bf16_t* p = base + (lane & 15) * LDP + ((lane >> 4) * 8);
    v8bf lo = *(const v8bf*)p;
    v8bf hi = *(const v8bf*)(p + 16);
    return __builtin_shufflevector(lo, hi, 0,1,2,3,4,5,6,7,8,9,10,11,12,13,14,15);
}

// ---------------------------------------------------------------------------
// Generic batched bf16 GEMM: D = act(scale*(A@B) + bias) + Res
//   A: [M,K] (or [K,M] when TRANS_A) bf16, B: [K,N] bf16
//   C (fp32) and/or Cb (bf16) outputs. Batch z -> (b1,b2) via bdim2 with
//   independent strides per operand (handles per-(batch,head) sub-matrices).
// Block: 128x128 tile, 256 threads = 8 waves (2x4), wave tile 32x64,
// 8x v_wmma_f32_16x16x32_bf16 per K-step per wave. Full A tiles staged by the
// Tensor Data Mover (one D# per wave, TENSORcnt-synchronized).
// ---------------------------------------------------------------------------
template<bool TRANS_A, bool BIAS, bool GELU_, bool RES>
__global__ __launch_bounds__(256, 2)
void gemm_bf16(const bf16_t* __restrict__ A, const bf16_t* __restrict__ B,
               float* __restrict__ C, bf16_t* __restrict__ Cb,
               const float* __restrict__ bias, const float* __restrict__ Res,
               int M, int N, int K,
               int lda, int ldb, int ldc, int ldcb,
               long sA1, long sA2, long sB1, long sB2,
               long sC1, long sC2, long sCb1, long sCb2,
               long sR1, long sR2, int bdim2, float scale)
{
    __shared__ __align__(16) bf16_t As[TILE][LDP];
    __shared__ __align__(16) bf16_t Bs[TILE][LDP];

    const int t    = threadIdx.x;
    const int lane = t & 31;
    const int wid  = t >> 5;
    const int wm   = (wid >> 1) * 32;   // 4 waves along M
    const int wn   = (wid & 1) * 64;    // 2 waves along N

    const int nblk = blockIdx.x, mblk = blockIdx.y, z = blockIdx.z;
    const int b1 = z / bdim2, b2 = z % bdim2;
    const long offA = (long)b1 * sA1 + (long)b2 * sA2;
    const long offB = (long)b1 * sB1 + (long)b2 * sB2;

    const v8f vzero = {0.f,0.f,0.f,0.f,0.f,0.f,0.f,0.f};
    v8f acc[2][4];
    #pragma unroll
    for (int i = 0; i < 2; ++i)
        #pragma unroll
        for (int j = 0; j < 4; ++j) acc[i][j] = vzero;

    const int KT = (K + KSTEP - 1) / KSTEP;
    for (int kt = 0; kt < KT; ++kt) {
        const int k0 = kt * KSTEP;
        // Full 128xKSTEP A tile in-range? (uniform predicate)
        const bool fullA = !TRANS_A && ((long)mblk * TILE + TILE <= (long)M)
                                    && (k0 + KSTEP <= K);
        // ---------------- stage A tile ----------------
        if (fullA) {
            // Tensor Data Mover: each wave DMAs its 16 rows (16x32 bf16),
            // hardware pads 64B rows to the 80B LDS pitch.
            const long r0 = (long)mblk * TILE + wid * 16;
            const unsigned lds = (unsigned)(uintptr_t)&As[wid * 16][0];
            tdm_load_tile_bf16(lds, A + offA + r0 * (long)lda + k0,
                               KSTEP, 16, (long)lda);
        } else if (!TRANS_A) {
            const int r = t >> 1, c = (t & 1) * 16;
            const long gr = (long)mblk * TILE + r;
            bf16_t* dst = &As[r][c];
            const bf16_t* src = A + offA + gr * (long)lda + (k0 + c);
            if (gr < M && (k0 + c + 16) <= K) {
                *(v8bf*)dst       = *(const v8bf*)src;
                *(v8bf*)(dst + 8) = *(const v8bf*)(src + 8);
            } else {
                #pragma unroll
                for (int i = 0; i < 16; ++i)
                    dst[i] = (gr < M && (k0 + c + i) < K) ? src[i] : (bf16_t)0.0f;
            }
        } else {
            const int kk = t >> 3, m0 = (t & 7) * 16;
            const long gk  = (long)k0 + kk;
            const long gm0 = (long)mblk * TILE + m0;
            if (gk < K && gm0 + 16 <= M) {
                const bf16_t* src = A + offA + gk * (long)lda + gm0;
                if (gk + KSTEP < K) __builtin_prefetch(src + (long)KSTEP * lda, 0, 1);
                v8bf lo = *(const v8bf*)src;
                v8bf hi = *(const v8bf*)(src + 8);
                #pragma unroll
                for (int i = 0; i < 8; ++i) { As[m0+i][kk] = lo[i]; As[m0+8+i][kk] = hi[i]; }
            } else {
                #pragma unroll
                for (int i = 0; i < 16; ++i) {
                    bf16_t v = (bf16_t)0.0f;
                    if (gk < K && gm0 + i < M) v = A[offA + gk * (long)lda + gm0 + i];
                    As[m0+i][kk] = v;
                }
            }
        }
        // ---------------- stage B tile (transposed to Bs[n][k]) ----------------
        {
            const int kk = t >> 3, n0 = (t & 7) * 16;
            const long gk  = (long)k0 + kk;
            const long gn0 = (long)nblk * TILE + n0;
            if (gk < K && gn0 + 16 <= N) {
                const bf16_t* src = B + offB + gk * (long)ldb + gn0;
                if (gk + KSTEP < K) __builtin_prefetch(src + (long)KSTEP * ldb, 0, 1);
                v8bf lo = *(const v8bf*)src;
                v8bf hi = *(const v8bf*)(src + 8);
                #pragma unroll
                for (int i = 0; i < 8; ++i) { Bs[n0+i][kk] = lo[i]; Bs[n0+8+i][kk] = hi[i]; }
            } else {
                #pragma unroll
                for (int i = 0; i < 16; ++i) {
                    bf16_t v = (bf16_t)0.0f;
                    if (gk < K && gn0 + i < N) v = B[offB + gk * (long)ldb + gn0 + i];
                    Bs[n0+i][kk] = v;
                }
            }
        }
        if (fullA) wait_tensorcnt0();   // wave's TDM done before barrier
        __syncthreads();
        // ---------------- compute: 8 WMMAs per wave per K-step ----------------
        v16bf afrag[2];
        #pragma unroll
        for (int mt = 0; mt < 2; ++mt) afrag[mt] = load_frag_a(&As[wm + mt*16][0], lane);
        #pragma unroll
        for (int nt = 0; nt < 4; ++nt) {
            v16bf bfrag = load_frag_b(&Bs[wn + nt*16][0], lane);
            #pragma unroll
            for (int mt = 0; mt < 2; ++mt)
                acc[mt][nt] = __builtin_amdgcn_wmma_f32_16x16x32_bf16(
                    false, afrag[mt], false, bfrag, (short)0, acc[mt][nt], false, false);
        }
        __syncthreads();
    }
    // ---------------- epilogue ----------------
    const long offC  = (long)b1 * sC1  + (long)b2 * sC2;
    const long offCb = (long)b1 * sCb1 + (long)b2 * sCb2;
    const long offR  = (long)b1 * sR1  + (long)b2 * sR2;
    const int rsub = (lane >> 4) * 8;   // D layout: M = v + 8*(lane>=16)
    const int ccol = lane & 15;         //           N = lane & 15
    #pragma unroll
    for (int mt = 0; mt < 2; ++mt)
        #pragma unroll
        for (int nt = 0; nt < 4; ++nt)
            #pragma unroll
            for (int v = 0; v < 8; ++v) {
                const long row = (long)mblk*TILE + wm + mt*16 + v + rsub;
                const long col = (long)nblk*TILE + wn + nt*16 + ccol;
                if (row < M && col < N) {
                    float x = acc[mt][nt][v] * scale;
                    if (BIAS)  x += bias[col];
                    if (GELU_) x = 0.5f * x * (1.0f + erff(x * 0.70710678118654752f));
                    if (RES)   x += Res[offR + row * (long)ldc + col];
                    if (C)  C [offC  + row * (long)ldc  + col] = x;
                    if (Cb) Cb[offCb + row * (long)ldcb + col] = (bf16_t)x;
                }
            }
}

// ---------------------------------------------------------------------------
// LayerNorm over rows of length 1024 (fp32 in -> bf16 out), one block per row
// ---------------------------------------------------------------------------
__global__ __launch_bounds__(256)
void ln_row(const float* __restrict__ X, const float* __restrict__ g,
            const float* __restrict__ bta, bf16_t* __restrict__ Y)
{
    const long row = blockIdx.x;
    const int  t   = threadIdx.x;
    const float* x = X + row * 1024L;
    float v[4], s = 0.f, ss = 0.f;
    #pragma unroll
    for (int i = 0; i < 4; ++i) { v[i] = x[t + 256*i]; s += v[i]; ss += v[i]*v[i]; }
    __shared__ float r1[256], r2[256];
    r1[t] = s; r2[t] = ss; __syncthreads();
    for (int o = 128; o; o >>= 1) {
        if (t < o) { r1[t] += r1[t+o]; r2[t] += r2[t+o]; }
        __syncthreads();
    }
    const float mean = r1[0] * (1.0f/1024.0f);
    const float var  = r2[0] * (1.0f/1024.0f) - mean*mean;
    const float rstd = rsqrtf(var + 1e-5f);
    bf16_t* y = Y + row * 1024L;
    #pragma unroll
    for (int i = 0; i < 4; ++i) {
        const int c = t + 256*i;
        y[c] = (bf16_t)(((v[i] - mean) * rstd) * g[c] + bta[c]);
    }
}

// ---------------------------------------------------------------------------
// Per-head LayerNorm (dim 64) applied in place to k or v third of bf16 QKV.
// One wave per row; wave32 shuffle reduction.
// ---------------------------------------------------------------------------
__global__ __launch_bounds__(256)
void ln_head(bf16_t* __restrict__ QKV, const float* __restrict__ g,
             const float* __restrict__ bta, int third, long nrows)
{
    const int lane = threadIdx.x & 31;
    const int wv   = threadIdx.x >> 5;
    const long row = (long)blockIdx.x * 8 + wv;      // row over B*N*HEADS
    if (row >= nrows) return;
    const long bn = row / 16;                        // HEADS = 16
    const int  h  = (int)(row % 16);
    bf16_t* p = QKV + bn * 3072L + (long)third * 1024 + h * 64;
    float a = (float)p[lane], b = (float)p[lane + 32];
    float s = a + b, ss = a*a + b*b;
    #pragma unroll
    for (int o = 16; o; o >>= 1) { s += __shfl_xor(s, o, 32); ss += __shfl_xor(ss, o, 32); }
    const float mean = s * (1.0f/64.0f);
    const float var  = ss * (1.0f/64.0f) - mean*mean;
    const float rstd = rsqrtf(var + 1e-5f);
    p[lane]      = (bf16_t)((a - mean) * rstd * g[lane]      + bta[lane]);
    p[lane + 32] = (bf16_t)((b - mean) * rstd * g[lane + 32] + bta[lane + 32]);
}

// ---------------------------------------------------------------------------
// BatchNorm over (B*N) per channel (PSI=64): one block per channel
// ---------------------------------------------------------------------------
__global__ __launch_bounds__(256)
void bn_stats(const float* __restrict__ Xp, float* __restrict__ stats, int rows)
{
    const int c = blockIdx.x, t = threadIdx.x;
    float s = 0.f, ss = 0.f;
    for (int r = t; r < rows; r += 256) {
        const float v = Xp[(long)r * 64 + c];
        s += v; ss += v*v;
    }
    __shared__ float r1[256], r2[256];
    r1[t] = s; r2[t] = ss; __syncthreads();
    for (int o = 128; o; o >>= 1) {
        if (t < o) { r1[t] += r1[t+o]; r2[t] += r2[t+o]; }
        __syncthreads();
    }
    if (t == 0) {
        const float m = r1[0] / rows;
        stats[c]      = m;
        stats[64 + c] = r2[0] / rows - m*m;
    }
}

__global__ void bn_norm(const float* __restrict__ Xp, const float* __restrict__ stats,
                        const float* __restrict__ g, const float* __restrict__ bta,
                        const float* __restrict__ mu,
                        bf16_t* __restrict__ Xb, bf16_t* __restrict__ Xsb, long total)
{
    const long i = (long)blockIdx.x * 256 + threadIdx.x;
    if (i >= total) return;
    const int c = (int)(i & 63);
    const float m = stats[c], var = stats[64 + c];
    const float v = (Xp[i] - m) * rsqrtf(var + 1e-5f) * g[c] + bta[c];
    const float mc = mu[c];
    const float sp = (mc > 20.f) ? mc : log1pf(expf(mc));   // softplus
    Xb[i]  = (bf16_t)v;
    Xsb[i] = (bf16_t)(v * sp);
}

// fp32 -> bf16 converter
__global__ void f2b(const float* __restrict__ X, bf16_t* __restrict__ Y, long n)
{
    const long i = (long)blockIdx.x * 256 + threadIdx.x;
    if (i < n) Y[i] = (bf16_t)X[i];
}

// ---------------------------------------------------------------------------
// Host-side GEMM dispatcher (selects template instantiation)
// ---------------------------------------------------------------------------
static void launch_gemm(hipStream_t s, bool TA, bool BIAS, bool GELU, bool RES,
    const bf16_t* A, const bf16_t* B, float* C, bf16_t* Cb,
    const float* bias, const float* Res,
    int M, int N, int K, int lda, int ldb, int ldc, int ldcb,
    long sA1, long sA2, long sB1, long sB2, long sC1, long sC2,
    long sCb1, long sCb2, long sR1, long sR2, int bdim2, int batches, float scale)
{
    dim3 grid((N + TILE - 1) / TILE, (M + TILE - 1) / TILE, batches);
    dim3 blk(256);
#define GO(ta,bi,ge,re) gemm_bf16<ta,bi,ge,re><<<grid,blk,0,s>>>(A,B,C,Cb,bias,Res,\
        M,N,K,lda,ldb,ldc,ldcb,sA1,sA2,sB1,sB2,sC1,sC2,sCb1,sCb2,sR1,sR2,bdim2,scale)
    if      (!TA && !BIAS && !GELU && !RES) GO(false,false,false,false);
    else if ( TA && !BIAS && !GELU && !RES) GO(true ,false,false,false);
    else if (!TA &&  BIAS && !GELU &&  RES) GO(false,true ,false,true );
    else if (!TA &&  BIAS &&  GELU && !RES) GO(false,true ,true ,false);
    else if (!TA &&  BIAS && !GELU && !RES) GO(false,true ,false,false);
    else if (!TA && !BIAS && !GELU &&  RES) GO(false,false,false,true );
#undef GO
}

// ---------------------------------------------------------------------------
// Pipeline
// ---------------------------------------------------------------------------
extern "C" void kernel_launch(void* const* d_in, const int* in_sizes, int n_in,
                              void* d_out, int out_size, void* d_ws, size_t ws_size,
                              hipStream_t stream)
{
    (void)in_sizes; (void)n_in; (void)out_size; (void)ws_size;
    const int  Bb = 4, NT = 4096, H = 1024, HEADS = 16, DH = 64, PSI = 64;
    const int  H3 = 3 * H, H4 = 4 * H;
    const long BN = (long)Bb * NT;                 // 16384

    const float* x    = (const float*)d_in[0];
    const float* fx   = (const float*)d_in[1];
    const float* mu   = (const float*)d_in[2];
    const float* ln1g = (const float*)d_in[3];
    const float* ln1b = (const float*)d_in[4];
    const float* Wqkv = (const float*)d_in[5];
    const float* kg   = (const float*)d_in[6];
    const float* kb_  = (const float*)d_in[7];
    const float* vg   = (const float*)d_in[8];
    const float* vb_  = (const float*)d_in[9];
    const float* Wout = (const float*)d_in[10];
    const float* bout = (const float*)d_in[11];
    const float* ln2g = (const float*)d_in[12];
    const float* ln2b = (const float*)d_in[13];
    const float* W1   = (const float*)d_in[14];
    const float* b1   = (const float*)d_in[15];
    const float* W2   = (const float*)d_in[16];
    const float* b2   = (const float*)d_in[17];
    const float* Wp   = (const float*)d_in[18];
    const float* bp   = (const float*)d_in[19];
    const float* bng  = (const float*)d_in[20];
    const float* bnb  = (const float*)d_in[21];
    const float* ln3g = (const float*)d_in[22];
    const float* ln3b = (const float*)d_in[23];
    const float* W3   = (const float*)d_in[24];
    const float* b3   = (const float*)d_in[25];
    const float* W4   = (const float*)d_in[26];
    const float* b4   = (const float*)d_in[27];

    // -------- workspace layout (sequentially reused regions) --------
    char* ws = (char*)d_ws;
    size_t off = 0;
    auto alloc = [&](size_t bytes) -> char* {
        char* p = ws + off;
        off = (off + bytes + 255) & ~(size_t)255;
        return p;
    };
    bf16_t* hb     = (bf16_t*)alloc((size_t)BN * H * 2);        // ln1/ln2/xb/fxb/ln3 (reused)
    bf16_t* wb     = (bf16_t*)alloc((size_t)H * H4 * 2);        // bf16 weights (reused)
    bf16_t* R1     = (bf16_t*)alloc((size_t)BN * H4 * 2);       // qkvb+attnb, then mid
    bf16_t* qkvb   = R1;
    bf16_t* attnb  = R1 + (size_t)BN * H3;
    bf16_t* mid    = R1;
    float*  dots   = (float*) alloc((size_t)Bb * HEADS * DH * DH * 4);
    bf16_t* dotsb  = (bf16_t*)alloc((size_t)Bb * HEADS * DH * DH * 2);
    float*  xbuf   = (float*) alloc((size_t)BN * H * 4);        // x after attn, then fx_new
    float*  xpsi   = (float*) alloc((size_t)BN * PSI * 4);
    bf16_t* xpsib  = (bf16_t*)alloc((size_t)BN * PSI * 2);
    bf16_t* xpsisb = (bf16_t*)alloc((size_t)BN * PSI * 2);
    float*  stats  = (float*) alloc((size_t)2 * PSI * 4);
    float*  kern   = (float*) alloc((size_t)Bb * PSI * H * 4);
    bf16_t* kernb  = (bf16_t*)alloc((size_t)Bb * PSI * H * 2);

    float* out_x  = (float*)d_out;
    float* out_fx = out_x + (size_t)BN * H;

    auto cvt = [&](const float* src, bf16_t* dst, long n) {
        f2b<<<dim3((unsigned)((n + 255) / 256)), dim3(256), 0, stream>>>(src, dst, n);
    };

    // ---- 1. h = LN1(x) -> bf16
    ln_row<<<dim3((unsigned)BN), dim3(256), 0, stream>>>(x, ln1g, ln1b, hb);

    // ---- 2. qkv = h @ Wqkv -> bf16 [BN, 3H]
    cvt(Wqkv, wb, (long)H * H3);
    launch_gemm(stream, false,false,false,false, hb, wb, nullptr, qkvb, nullptr, nullptr,
                (int)BN, H3, H, H, H3, 0, H3,
                0,0, 0,0, 0,0, 0,0, 0,0, 1, 1, 1.0f);

    // ---- 3. per-head LN of k and v (in place, bf16)
    {
        const long nrows = BN * HEADS;
        dim3 g((unsigned)(nrows / 8));
        ln_head<<<g, dim3(256), 0, stream>>>(qkvb, kg, kb_, 1, nrows);
        ln_head<<<g, dim3(256), 0, stream>>>(qkvb, vg, vb_, 2, nrows);
    }

    // ---- 4. dots[b,h] = k^T v  (M=64,N=64,K=4096, 64 batches, TRANS_A)
    launch_gemm(stream, true,false,false,false,
                qkvb + H, qkvb + 2*H, dots, nullptr, nullptr, nullptr,
                DH, DH, NT, H3, H3, DH, 0,
                (long)NT*H3, DH, (long)NT*H3, DH,
                (long)HEADS*DH*DH, (long)DH*DH, 0,0, 0,0,
                HEADS, Bb*HEADS, 1.0f);
    cvt(dots, dotsb, (long)Bb*HEADS*DH*DH);

    // ---- 5. attn[b,h] = (q @ dots) / N -> bf16 [B,N,H]
    launch_gemm(stream, false,false,false,false,
                qkvb, dotsb, nullptr, attnb, nullptr, nullptr,
                NT, DH, DH, H3, DH, 0, H,
                (long)NT*H3, DH, (long)HEADS*DH*DH, (long)DH*DH,
                0,0, (long)NT*H, DH, 0,0,
                HEADS, Bb*HEADS, 1.0f/(float)NT);

    // ---- 6. x = attn @ Wout + bout + x
    cvt(Wout, wb, (long)H * H);
    launch_gemm(stream, false,true,false,true,
                attnb, wb, xbuf, nullptr, bout, x,
                (int)BN, H, H, H, H, H, 0,
                0,0, 0,0, 0,0, 0,0, 0,0, 1, 1, 1.0f);

    // ---- 7. MLP: x = GELU(LN2(x)@W1+b1)@W2 + b2 + x   (x final -> d_out)
    ln_row<<<dim3((unsigned)BN), dim3(256), 0, stream>>>(xbuf, ln2g, ln2b, hb);
    cvt(W1, wb, (long)H * H4);
    launch_gemm(stream, false,true,true,false,
                hb, wb, nullptr, mid, b1, nullptr,
                (int)BN, H4, H, H, H4, 0, H4,
                0,0, 0,0, 0,0, 0,0, 0,0, 1, 1, 1.0f);
    cvt(W2, wb, (long)H4 * H);
    launch_gemm(stream, false,true,false,true,
                mid, wb, out_x, nullptr, b2, xbuf,
                (int)BN, H, H4, H4, H, H, 0,
                0,0, 0,0, 0,0, 0,0, 0,0, 1, 1, 1.0f);

    // ---- 8. psi projection + BatchNorm
    cvt(out_x, hb, BN * H);            // x final -> bf16
    cvt(Wp, wb, (long)H * PSI);
    launch_gemm(stream, false,true,false,false,
                hb, wb, xpsi, nullptr, bp, nullptr,
                (int)BN, PSI, H, H, PSI, PSI, 0,
                0,0, 0,0, 0,0, 0,0, 0,0, 1, 1, 1.0f);
    bn_stats<<<dim3(PSI), dim3(256), 0, stream>>>(xpsi, stats, (int)BN);
    bn_norm<<<dim3((unsigned)((BN*PSI + 255)/256)), dim3(256), 0, stream>>>(
        xpsi, stats, bng, bnb, mu, xpsib, xpsisb, BN*PSI);

    // ---- 9. kern[b] = x_^T @ fx   (M=64,N=1024,K=4096, TRANS_A, 4 batches)
    cvt(fx, hb, BN * H);               // fx -> bf16
    launch_gemm(stream, true,false,false,false,
                xpsib, hb, kern, nullptr, nullptr, nullptr,
                PSI, H, NT, PSI, H, H, 0,
                (long)NT*PSI, 0, (long)NT*H, 0,
                (long)PSI*H, 0, 0,0, 0,0, 1, Bb, 1.0f);
    cvt(kern, kernb, (long)Bb*PSI*H);

    // ---- 10. fx_new = (x_*softplus(mu)) @ kern + fx   -> xbuf
    launch_gemm(stream, false,false,false,true,
                xpsisb, kernb, xbuf, nullptr, nullptr, fx,
                NT, H, PSI, PSI, H, H, 0,
                (long)NT*PSI, 0, (long)PSI*H, 0,
                (long)NT*H, 0, 0,0, (long)NT*H, 0, 1, Bb, 1.0f);

    // ---- 11. mlp2: fx = GELU(LN3(fx_new)@W3+b3)@W4 + b4  -> d_out (fx half)
    ln_row<<<dim3((unsigned)BN), dim3(256), 0, stream>>>(xbuf, ln3g, ln3b, hb);
    cvt(W3, wb, (long)H * H4);
    launch_gemm(stream, false,true,true,false,
                hb, wb, nullptr, mid, b3, nullptr,
                (int)BN, H4, H, H, H4, 0, H4,
                0,0, 0,0, 0,0, 0,0, 0,0, 1, 1, 1.0f);
    cvt(W4, wb, (long)H4 * H);
    launch_gemm(stream, false,true,false,false,
                mid, wb, out_fx, nullptr, b4, nullptr,
                (int)BN, H, H4, H4, H, H, 0,
                0,0, 0,0, 0,0, 0,0, 0,0, 1, 1, 1.0f);
}
